// PointNetFeaturePropagation_81836306858516
// MI455X (gfx1250) — matrix operand order repaired and checked by reading, not compiled, not run on hardware
//
#include <hip/hip_runtime.h>
#include <hip/hip_bf16.h>

// ---------------------------------------------------------------------------
// Types for CDNA5 WMMA (wave32)
// ---------------------------------------------------------------------------
typedef __attribute__((ext_vector_type(16))) __bf16 v16bf;
typedef __attribute__((ext_vector_type(8)))  float  v8f;
typedef __attribute__((ext_vector_type(4)))  int    v4i;

#define MTOT   32768   // B*N = 4*8192
#define NPTS   8192
#define SPTS   2048
#define BATCH  4
#define D1     128
#define D2     256
#define CIN    384
#define COUT   256

union FragU { v4i i[2]; v16bf v; };

__device__ __forceinline__ v16bf load_frag(const __bf16* p) {
  FragU f;
  f.i[0] = *(const v4i*)(p);
  f.i[1] = *(const v4i*)(p + 8);
  return f.v;
}

__device__ __forceinline__ unsigned pack2bf(float a, float b) {
  union { __bf16 h[2]; unsigned u; } q;
  q.h[0] = (__bf16)a;
  q.h[1] = (__bf16)b;
  return q.u;
}

// B-fragment packed layout (bf16 32x16 KxN tiles, wave32 lane order):
//   lane n (<16): col N=n, elems e=K 0..15 ; lane n+16: col N=n, e=K 16..31
// element offset = ((tile_m*KT + tile_k)*32 + lane)*16 + e
__device__ __forceinline__ void store_packB_pair(__bf16* base, int KT, int c,
                                                 int m, float a, float b) {
  int tile_m = m >> 4, nn = m & 15;
  int tile_k = c >> 5, kk = c & 31;
  int lane = nn + ((kk >> 4) << 4);
  int e = kk & 15;                       // even (c is even)
  size_t off = (((size_t)tile_m * KT + tile_k) << 9) + (lane << 4) + e;
  *(unsigned*)(base + off) = pack2bf(a, b);
}

// ---------------------------------------------------------------------------
// Kernel 1: KNN(k=3) + inverse-distance interpolation + concat + bf16 pack
// ---------------------------------------------------------------------------
__global__ void __launch_bounds__(128)
knn_interp_pack(const float* __restrict__ xyz1, const float* __restrict__ xyz2,
                const float* __restrict__ points1,
                const float* __restrict__ points2, __bf16* __restrict__ Xpack) {
  __shared__ float sx[SPTS], sy[SPTS], sz[SPTS];
  const int b = blockIdx.y;
  const int n = blockIdx.x * 128 + threadIdx.x;

  for (int s = threadIdx.x; s < SPTS; s += 128) {
    const float* p = xyz2 + ((size_t)b * SPTS + s) * 3;
    sx[s] = p[0]; sy[s] = p[1]; sz[s] = p[2];
  }
  __syncthreads();

  const float* q = xyz1 + ((size_t)b * NPTS + n) * 3;
  const float qx = q[0], qy = q[1], qz = q[2];

  float d0 = 3.4e38f, d1 = 3.4e38f, d2 = 3.4e38f;
  int   i0 = 0, i1 = 0, i2 = 0;
  for (int s = 0; s < SPTS; ++s) {
    float dx = qx - sx[s], dy = qy - sy[s], dz = qz - sz[s];
    float d = fmaf(dx, dx, fmaf(dy, dy, dz * dz));
    if (d < d0)      { d2 = d1; i2 = i1; d1 = d0; i1 = i0; d0 = d; i0 = s; }
    else if (d < d1) { d2 = d1; i2 = i1; d1 = d; i1 = s; }
    else if (d < d2) { d2 = d; i2 = s; }
  }
  float r0 = 1.0f / (d0 + 1e-8f);
  float r1 = 1.0f / (d1 + 1e-8f);
  float r2 = 1.0f / (d2 + 1e-8f);
  float inv = 1.0f / (r0 + r1 + r2);
  float w0 = r0 * inv, w1 = r1 * inv, w2 = r2 * inv;

  const int m = b * NPTS + n;
  const int KT = CIN >> 5;  // 12

  // channels [0,128): skip features from points1 [B,D1,N]
  const float* p1 = points1 + (size_t)b * D1 * NPTS + n;
  for (int c = 0; c < D1; c += 2) {
    float v0 = p1[(size_t)c * NPTS];
    float v1 = p1[(size_t)(c + 1) * NPTS];
    store_packB_pair(Xpack, KT, c, m, v0, v1);
  }
  // channels [128,384): interpolated points2 [B,D2,S]
  const float* p2 = points2 + (size_t)b * D2 * SPTS;
  for (int c = 0; c < D2; c += 2) {
    const float* r  = p2 + (size_t)c * SPTS;
    const float* r1p = r + SPTS;
    float v0 = w0 * r[i0]  + w1 * r[i1]  + w2 * r[i2];
    float v1 = w0 * r1p[i0] + w1 * r1p[i1] + w2 * r1p[i2];
    store_packB_pair(Xpack, KT, D1 + c, m, v0, v1);
  }
}

// ---------------------------------------------------------------------------
// Kernel 2: pack weights [O,C] f32 -> A-fragment bf16 (16x32 MxK tiles)
//   lane<16: M=lane, K=e (0..15)
//   lane>=16: M=lane-16, K=16+e (16..31)
// ---------------------------------------------------------------------------
__global__ void pack_weights(const float* __restrict__ W,
                             __bf16* __restrict__ Apack, int O, int C) {
  const int KT = C >> 5;
  int idx = blockIdx.x * 256 + threadIdx.x;
  if (idx >= O * C) return;
  int e = idx & 15;
  int lane = (idx >> 4) & 31;
  int t = idx >> 9;             // t = tile_o*KT + kt
  int kt = t % KT;
  int to = t / KT;
  int K = kt * 32 + e + ((lane >> 4) << 4);
  int M = to * 16 + (lane & 15);
  Apack[idx] = (__bf16)W[(size_t)M * C + K];
}

// ---------------------------------------------------------------------------
// Kernel 3: GEMM  Y[O,MTOT] = A[O,C] * B[C,MTOT]   (bf16 in, f32 out)
// block = 256 thr = 8 waves, block tile 64(O) x 128(M), wave tile 32x32.
// ---------------------------------------------------------------------------
__global__ void __launch_bounds__(256)
gemm_wmma(const __bf16* __restrict__ Apack, const __bf16* __restrict__ Bpack,
          float* __restrict__ Y, int KT) {
  const int lane = threadIdx.x & 31;
  const int wave = threadIdx.x >> 5;
  const int o0 = blockIdx.y * 64 + (wave & 1) * 32;
  const int m0 = blockIdx.x * 128 + (wave >> 1) * 32;

  const __bf16* aP = Apack + (((size_t)(o0 >> 4) * KT) << 9) + (lane << 4);
  const __bf16* bP = Bpack + (((size_t)(m0 >> 4) * KT) << 9) + (lane << 4);
  const size_t tileStride = (size_t)KT << 9;  // next 16-row/col tile group

  v8f acc00 = {}, acc01 = {}, acc10 = {}, acc11 = {};

  for (int kt = 0; kt < KT; ++kt) {
    const size_t ko = (size_t)kt << 9;
    v16bf a0 = load_frag(aP + ko);
    v16bf a1 = load_frag(aP + tileStride + ko);
    v16bf b0 = load_frag(bP + ko);
    v16bf b1 = load_frag(bP + tileStride + ko);
    if (kt + 1 < KT) {  // hint next K tile into cache (global_prefetch_b8)
      __builtin_prefetch(bP + ko + 512, 0, 0);
      __builtin_prefetch(aP + ko + 512, 0, 0);
    }
    acc00 = __builtin_amdgcn_wmma_f32_16x16x32_bf16(false, a0, false, b0,
                                                    (short)0, acc00, false, false);
    acc01 = __builtin_amdgcn_wmma_f32_16x16x32_bf16(false, a0, false, b1,
                                                    (short)0, acc01, false, false);
    acc10 = __builtin_amdgcn_wmma_f32_16x16x32_bf16(false, a1, false, b0,
                                                    (short)0, acc10, false, false);
    acc11 = __builtin_amdgcn_wmma_f32_16x16x32_bf16(false, a1, false, b1,
                                                    (short)0, acc11, false, false);
  }

  // C/D layout: lane<16 -> (M=r, N=lane); lane>=16 -> (M=8+r, N=lane-16)
  const int mcol = lane & 15;
  const int rbase = (lane >> 4) << 3;
  for (int r = 0; r < 8; ++r) {
    int orow = o0 + rbase + r;
    float* y0 = Y + (size_t)orow * MTOT;
    float* y1 = Y + (size_t)(orow + 16) * MTOT;
    y0[m0 + mcol]      = acc00[r];
    y0[m0 + 16 + mcol] = acc01[r];
    y1[m0 + mcol]      = acc10[r];
    y1[m0 + 16 + mcol] = acc11[r];
  }
}

// ---------------------------------------------------------------------------
// Kernel 4: per-channel BN statistics -> scale/shift
// (conv bias cancels exactly under training-mode BN, so it is omitted)
// ---------------------------------------------------------------------------
__global__ void __launch_bounds__(256)
bn_stats(const float* __restrict__ Y, const float* __restrict__ gamma,
         const float* __restrict__ beta, float* __restrict__ scale,
         float* __restrict__ shift) {
  __shared__ float ssum[256], ssq[256];
  const int c = blockIdx.x;
  const int tid = threadIdx.x;
  const float* row = Y + (size_t)c * MTOT;
  float s = 0.f, q = 0.f;
  for (int m = tid; m < MTOT; m += 256) {
    float v = row[m];
    s += v;
    q = fmaf(v, v, q);
  }
  ssum[tid] = s; ssq[tid] = q;
  __syncthreads();
  for (int st = 128; st > 0; st >>= 1) {
    if (tid < st) { ssum[tid] += ssum[tid + st]; ssq[tid] += ssq[tid + st]; }
    __syncthreads();
  }
  if (tid == 0) {
    float mean = ssum[0] * (1.0f / MTOT);
    float var = ssq[0] * (1.0f / MTOT) - mean * mean;
    float sc = gamma[c] * rsqrtf(var + 1e-5f);
    scale[c] = sc;
    shift[c] = beta[c] - mean * sc;
  }
}

// ---------------------------------------------------------------------------
// Kernel 5: apply BN(+ReLU); optionally save f32 copy (residual); repack bf16
// one thread per (channel-pair, m)
// ---------------------------------------------------------------------------
__global__ void __launch_bounds__(256)
bn_apply_pack(const float* __restrict__ Y, const float* __restrict__ scale,
              const float* __restrict__ shift, float* __restrict__ Xres,
              __bf16* __restrict__ Bpack, int KT) {
  int idx = blockIdx.x * 256 + threadIdx.x;
  int m = idx & (MTOT - 1);
  int c = (idx >> 15) << 1;
  float v0 = fmaf(Y[(size_t)c * MTOT + m], scale[c], shift[c]);
  float v1 = fmaf(Y[(size_t)(c + 1) * MTOT + m], scale[c + 1], shift[c + 1]);
  v0 = fmaxf(v0, 0.f);
  v1 = fmaxf(v1, 0.f);
  if (Xres) {
    Xres[(size_t)c * MTOT + m] = v0;
    Xres[(size_t)(c + 1) * MTOT + m] = v1;
  }
  store_packB_pair(Bpack, KT, c, m, v0, v1);
}

// ---------------------------------------------------------------------------
// Kernel 6: final BN (no relu) + residual + relu, remap [O,M] -> [B,C,N]
// ---------------------------------------------------------------------------
__global__ void __launch_bounds__(256)
final_residual(const float* __restrict__ Y2, const float* __restrict__ scale,
               const float* __restrict__ shift, const float* __restrict__ Xres,
               float* __restrict__ out) {
  int idx = blockIdx.x * 256 + threadIdx.x;  // idx = c*MTOT + m
  int m = idx & (MTOT - 1);
  int c = idx >> 15;
  float v = fmaf(Y2[idx], scale[c], shift[c]) + Xres[idx];
  v = fmaxf(v, 0.f);
  int b = m >> 13;
  int n = m & (NPTS - 1);
  out[((size_t)b * COUT + c) * NPTS + n] = v;
}

// ---------------------------------------------------------------------------
// Host-side launcher
// ---------------------------------------------------------------------------
extern "C" void kernel_launch(void* const* d_in, const int* in_sizes, int n_in,
                              void* d_out, int out_size, void* d_ws,
                              size_t ws_size, hipStream_t stream) {
  const float* xyz1    = (const float*)d_in[0];
  const float* xyz2    = (const float*)d_in[1];
  const float* points1 = (const float*)d_in[2];
  const float* points2 = (const float*)d_in[3];
  const float* fuse_w  = (const float*)d_in[4];
  const float* fuse_g  = (const float*)d_in[6];
  const float* fuse_be = (const float*)d_in[7];
  const float* e1_w    = (const float*)d_in[8];
  const float* e1_g    = (const float*)d_in[10];
  const float* e1_be   = (const float*)d_in[11];
  const float* e2_w    = (const float*)d_in[12];
  const float* e2_g    = (const float*)d_in[14];
  const float* e2_be   = (const float*)d_in[15];
  float* out = (float*)d_out;

  auto align256 = [](size_t x) { return (x + 255) & ~(size_t)255; };
  char* ws = (char*)d_ws;
  __bf16* Xpack0 = (__bf16*)ws; ws += align256((size_t)CIN * MTOT * 2);
  __bf16* XpackN = (__bf16*)ws; ws += align256((size_t)COUT * MTOT * 2);
  float*  Ybuf   = (float*)ws;  ws += align256((size_t)COUT * MTOT * 4);
  float*  Xres   = (float*)ws;  ws += align256((size_t)COUT * MTOT * 4);
  __bf16* WpF    = (__bf16*)ws; ws += align256((size_t)COUT * CIN * 2);
  __bf16* Wp1    = (__bf16*)ws; ws += align256((size_t)COUT * COUT * 2);
  __bf16* Wp2    = (__bf16*)ws; ws += align256((size_t)COUT * COUT * 2);
  float*  scl    = (float*)ws;  // 6*256 floats: scale0,shift0,scale1,...
  float* scale0 = scl,          *shift0 = scl + 256;
  float* scale1 = scl + 512,    *shift1 = scl + 768;
  float* scale2 = scl + 1024,   *shift2 = scl + 1280;

  // 1) KNN + interpolate + concat + pack
  knn_interp_pack<<<dim3(NPTS / 128, BATCH), 128, 0, stream>>>(
      xyz1, xyz2, points1, points2, Xpack0);

  // 2) pack the three weight matrices
  pack_weights<<<(COUT * CIN + 255) / 256, 256, 0, stream>>>(fuse_w, WpF, COUT, CIN);
  pack_weights<<<(COUT * COUT + 255) / 256, 256, 0, stream>>>(e1_w, Wp1, COUT, COUT);
  pack_weights<<<(COUT * COUT + 255) / 256, 256, 0, stream>>>(e2_w, Wp2, COUT, COUT);

  const dim3 ggrid(MTOT / 128, COUT / 64);

  // fuse: Y0 = fuse_w * X0 ; BN+ReLU -> Xres (f32) + Xpack1
  gemm_wmma<<<ggrid, 256, 0, stream>>>(WpF, Xpack0, Ybuf, CIN / 32);
  bn_stats<<<COUT, 256, 0, stream>>>(Ybuf, fuse_g, fuse_be, scale0, shift0);
  bn_apply_pack<<<(COUT / 2) * MTOT / 256, 256, 0, stream>>>(
      Ybuf, scale0, shift0, Xres, XpackN, COUT / 32);

  // e1: Y1 = e1_w * X1 ; BN+ReLU -> Xpack2
  gemm_wmma<<<ggrid, 256, 0, stream>>>(Wp1, XpackN, Ybuf, COUT / 32);
  bn_stats<<<COUT, 256, 0, stream>>>(Ybuf, e1_g, e1_be, scale1, shift1);
  bn_apply_pack<<<(COUT / 2) * MTOT / 256, 256, 0, stream>>>(
      Ybuf, scale1, shift1, nullptr, XpackN, COUT / 32);

  // e2: Y2 = e2_w * X2 ; BN (no relu) + residual + relu -> out
  gemm_wmma<<<ggrid, 256, 0, stream>>>(Wp2, XpackN, Ybuf, COUT / 32);
  bn_stats<<<COUT, 256, 0, stream>>>(Ybuf, e2_g, e2_be, scale2, shift2);
  final_residual<<<COUT * MTOT / 256, 256, 0, stream>>>(
      Ybuf, scale2, shift2, Xres, out);
}
